// NSTLoss_6768868459169
// MI455X (gfx1250) — compile-verified, matrix operand order checked by reading, not run on hardware
//
#include <hip/hip_runtime.h>
#include <hip/hip_bf16.h>

typedef __bf16 bf16_t;
typedef __attribute__((ext_vector_type(16))) __bf16 v16bf;
typedef __attribute__((ext_vector_type(8)))  __bf16 v8bf;
typedef __attribute__((ext_vector_type(8)))  float  v8f;

#define BB   16
#define CC   512
#define NROW 1024            // 2*C  (fs rows then ft rows)
#define DD   1024            // spatial dim after pooling (32*32)
#define TILE 128             // output tile per workgroup
#define KCH  32              // K per WMMA step (bf16)
#define NKC  (DD / KCH)      // 32 K-chunks
#define LDS_STRIDE 80        // bytes per LDS row: 64 data + 16 pad (16B aligned, conflict-free)
#define STRIP (TILE * LDS_STRIDE)

// ---- CDNA5 async Global->LDS DMA (bypasses VGPRs, tracked by ASYNCcnt) ----
__device__ __forceinline__ void async_b128(unsigned lds_off, const char* gptr) {
    asm volatile("global_load_async_to_lds_b128 %0, %1, off"
                 :: "v"(lds_off), "v"(gptr) : "memory");
}
__device__ __forceinline__ void wait_async0() {
#if defined(__has_builtin) && __has_builtin(__builtin_amdgcn_s_wait_asynccnt)
    __builtin_amdgcn_s_wait_asynccnt(0);
#else
    asm volatile("s_wait_asynccnt 0x0" ::: "memory");
#endif
}

// ---------------------------------------------------------------------------
// Phase 1: pool (fs) / flatten (ft), L2-normalize rows, emit bf16 U rows.
// One workgroup per (batch, row). Row < 512 -> pooled f_s, else f_t.
// ---------------------------------------------------------------------------
__global__ __launch_bounds__(256) void nst_preprocess(
    const float* __restrict__ fs, const float* __restrict__ ft,
    bf16_t* __restrict__ U)
{
    const int blk  = blockIdx.x;
    const int b    = blk >> 10;
    const int row  = blk & 1023;
    const int t    = threadIdx.x;
    const int lane = t & 31;
    const int w    = t >> 5;

    float v[4];
    if (row < CC) {
        // 2x2 average pool of a 64x64 image -> 1024 values
        const float2* src = (const float2*)(fs + (size_t)(b * CC + row) * 4096);
        #pragma unroll
        for (int i = 0; i < 4; ++i) {
            int d = t + i * 256;
            int y = d >> 5, x = d & 31;
            float2 a0 = src[(size_t)64 * y + x];        // row 2y, cols 2x,2x+1
            float2 a1 = src[(size_t)64 * y + x + 32];   // row 2y+1
            v[i] = 0.25f * (a0.x + a0.y + a1.x + a1.y);
        }
    } else {
        const float* src = ft + (size_t)(b * CC + (row - CC)) * 1024;
        #pragma unroll
        for (int i = 0; i < 4; ++i) v[i] = src[t + i * 256];
    }

    float ss = v[0]*v[0] + v[1]*v[1] + v[2]*v[2] + v[3]*v[3];
    #pragma unroll
    for (int off = 16; off; off >>= 1) ss += __shfl_xor(ss, off, 32);

    __shared__ float wsum[8];
    if (lane == 0) wsum[w] = ss;
    __syncthreads();
    float tot = wsum[0] + wsum[1] + wsum[2] + wsum[3]
              + wsum[4] + wsum[5] + wsum[6] + wsum[7];
    float scale = 1.0f / fmaxf(sqrtf(tot), 1e-12f);

    bf16_t* dst = U + ((size_t)b * NROW + row) * DD;
    #pragma unroll
    for (int i = 0; i < 4; ++i) dst[t + i * 256] = (bf16_t)(v[i] * scale);
}

// ---------------------------------------------------------------------------
// Phase 2: per-batch G = U·Uᵀ (1024x1024, K=1024) with bf16 WMMA, fused
// sign-weighted sum of G² (+1 diag 512-blocks, -1 off-diag). One 128x128
// output tile per workgroup; 8 waves, each a 32x64 sub-tile (2x4 WMMAs).
// Double-buffered async Global->LDS DMA: one barrier + one asynccnt wait
// per K-chunk; DMA for chunk k+1 overlaps WMMAs of chunk k.
// ---------------------------------------------------------------------------
__device__ __forceinline__ void issue_chunk(const char* gA, const char* gB,
                                            unsigned dA, unsigned dB, int k0) {
    const char* a = gA + (size_t)k0 * 2;
    const char* b = gB + (size_t)k0 * 2;
    async_b128(dA,      a);
    async_b128(dA + 16, a + 16);
    async_b128(dB,      b);
    async_b128(dB + 16, b + 16);
}

__device__ __forceinline__ void wmma_step(const unsigned char* bA,
                                          const unsigned char* bB,
                                          int wr, int wc, int fl, int fh,
                                          v8f acc[2][4]) {
    v16bf af[2], bfr[4];
    #pragma unroll
    for (int mi = 0; mi < 2; ++mi) {
        int r = wr * 32 + mi * 16 + fl;
        v8bf lo = *(const v8bf*)(bA + r * LDS_STRIDE + fh);
        v8bf hi = *(const v8bf*)(bA + r * LDS_STRIDE + fh + 32);
        af[mi] = __builtin_shufflevector(lo, hi,
                   0,1,2,3,4,5,6,7,8,9,10,11,12,13,14,15);
    }
    #pragma unroll
    for (int ni = 0; ni < 4; ++ni) {
        int r = wc * 64 + ni * 16 + fl;
        v8bf lo = *(const v8bf*)(bB + r * LDS_STRIDE + fh);
        v8bf hi = *(const v8bf*)(bB + r * LDS_STRIDE + fh + 32);
        bfr[ni] = __builtin_shufflevector(lo, hi,
                    0,1,2,3,4,5,6,7,8,9,10,11,12,13,14,15);
    }
    #pragma unroll
    for (int mi = 0; mi < 2; ++mi)
        #pragma unroll
        for (int ni = 0; ni < 4; ++ni)
            acc[mi][ni] = __builtin_amdgcn_wmma_f32_16x16x32_bf16(
                false, af[mi], false, bfr[ni],
                (short)0, acc[mi][ni], false, false);
}

__global__ __launch_bounds__(256) void nst_gram(
    const bf16_t* __restrict__ U, float* __restrict__ partials)
{
    __shared__ __align__(16) unsigned char ldsbuf[2][2 * STRIP]; // [buf][A strip | B strip]
    __shared__ float wpart[8];

    const int blk  = blockIdx.x;
    const int b    = blk >> 6;
    const int ti   = (blk >> 3) & 7;
    const int tj   = blk & 7;
    const int t    = threadIdx.x;
    const int lane = t & 31;
    const int w    = t >> 5;
    const int wr   = w >> 1;   // 0..3 : 32-row strip within tile
    const int wc   = w & 1;    // 0..1 : 64-col strip within tile

    const char* Ub = (const char*)(U + (size_t)b * NROW * DD);

    // staging: thread t owns 32 bytes of row (t>>1), byte offset (t&1)*32
    const int srow = t >> 1;
    const int soff = (t & 1) * 32;
    const char* gA0 = Ub + ((size_t)(ti * TILE + srow)) * (DD * 2) + soff;
    const char* gB0 = Ub + ((size_t)(tj * TILE + srow)) * (DD * 2) + soff;

    // per-thread DMA destination LDS offsets (low 32 bits of flat LDS ptr)
    unsigned dA[2], dB[2];
    #pragma unroll
    for (int c = 0; c < 2; ++c) {
        dA[c] = (unsigned)(size_t)&ldsbuf[c][srow * LDS_STRIDE + soff];
        dB[c] = (unsigned)(size_t)&ldsbuf[c][STRIP + srow * LDS_STRIDE + soff];
    }

    // per-lane fragment addressing (16-bit A 16x32 ISA layout):
    // lane<16 holds K{0..7,16..23} (bytes [0,16)+[32,48)), lane>=16 the rest
    const int fl = lane & 15;
    const int fh = (lane >> 4) * 16;

    v8f acc[2][4];
    #pragma unroll
    for (int mi = 0; mi < 2; ++mi)
        #pragma unroll
        for (int ni = 0; ni < 4; ++ni)
            acc[mi][ni] = (v8f){0.f,0.f,0.f,0.f,0.f,0.f,0.f,0.f};

    // prologue: DMA chunk 0 into buffer 0
    issue_chunk(gA0, gB0, dA[0], dB[0], 0);

    #pragma unroll 1
    for (int kc = 0; kc < NKC; kc += 2) {
        // ---- even chunk: compute buf0, stream kc+1 into buf1 ----
        wait_async0();         // this wave's DMA into buf0 complete
        __syncthreads();       // all waves' DMA done + all prior buf1 reads done
        issue_chunk(gA0, gB0, dA[1], dB[1], (kc + 1) * KCH);
        wmma_step(ldsbuf[0], ldsbuf[0] + STRIP, wr, wc, fl, fh, acc);

        // ---- odd chunk: compute buf1, stream kc+2 into buf0 ----
        wait_async0();
        __syncthreads();
        if (kc + 2 < NKC)
            issue_chunk(gA0, gB0, dA[0], dB[0], (kc + 2) * KCH);
        wmma_step(ldsbuf[1], ldsbuf[1] + STRIP, wr, wc, fl, fh, acc);
    }

    // sign-weighted sum of squares: +1 if (i<512)==(j<512), else -1.
    float part = 0.0f;
    #pragma unroll
    for (int mi = 0; mi < 2; ++mi) {
        int gi = ti * TILE + wr * 32 + mi * 16;
        #pragma unroll
        for (int ni = 0; ni < 4; ++ni) {
            int gj = tj * TILE + wc * 64 + ni * 16;
            float s = 0.0f;
            #pragma unroll
            for (int e = 0; e < 8; ++e) s += acc[mi][ni][e] * acc[mi][ni][e];
            part += ((gi < CC) == (gj < CC)) ? s : -s;
        }
    }
    #pragma unroll
    for (int off = 16; off; off >>= 1) part += __shfl_xor(part, off, 32);
    if (lane == 0) wpart[w] = part;
    __syncthreads();
    if (t == 0)
        partials[blk] = wpart[0] + wpart[1] + wpart[2] + wpart[3]
                      + wpart[4] + wpart[5] + wpart[6] + wpart[7];
}

// ---------------------------------------------------------------------------
// Phase 3: deterministic fixed-order reduction of 1024 partials -> scalar.
// ---------------------------------------------------------------------------
__global__ __launch_bounds__(256) void nst_reduce(
    const float* __restrict__ partials, float* __restrict__ out)
{
    __shared__ float sm[256];
    int t = threadIdx.x;
    sm[t] = partials[t] + partials[t + 256] + partials[t + 512] + partials[t + 768];
    __syncthreads();
    #pragma unroll
    for (int off = 128; off; off >>= 1) {
        if (t < off) sm[t] += sm[t + off];
        __syncthreads();
    }
    if (t == 0) out[0] = sm[0] * (1.0f / 4194304.0f);  // 1/(B*C*C)
}

extern "C" void kernel_launch(void* const* d_in, const int* in_sizes, int n_in,
                              void* d_out, int out_size, void* d_ws, size_t ws_size,
                              hipStream_t stream) {
    (void)in_sizes; (void)n_in; (void)out_size; (void)ws_size;
    const float* fs = (const float*)d_in[0];  // (16,512,64,64)
    const float* ft = (const float*)d_in[1];  // (16,512,32,32)
    float* out = (float*)d_out;

    bf16_t* U        = (bf16_t*)d_ws;                                   // 32 MB bf16
    float*  partials = (float*)((char*)d_ws + (size_t)BB * NROW * DD * sizeof(bf16_t));

    nst_preprocess<<<BB * NROW, 256, 0, stream>>>(fs, ft, U);
    nst_gram      <<<BB * 64,   256, 0, stream>>>(U, partials);
    nst_reduce    <<<1,         256, 0, stream>>>(partials, out);
}
